// TransformerBlock_64132451663945
// MI455X (gfx1250) — matrix operand order repaired and checked by reading, not compiled, not run on hardware
//
#include <hip/hip_runtime.h>
#include <hip/hip_bf16.h>
#include <cstdint>
#include <cstddef>

#define NTOK    16384     // B*S = 4*4096
#define DMODEL  1024
#define NHEADS  16
#define DK      64
#define DFF     4096
#define LN_EPS  1e-5f
#define APAD    72        // padded LDS row (halfs): 144B = 9x16B, bank-conflict-free

typedef __attribute__((ext_vector_type(16))) __bf16 v16bf;
typedef __attribute__((ext_vector_type(8)))  float  v8f;
typedef __attribute__((ext_vector_type(2)))  float  v2f;

union FragBF { uint4 q[2]; v16bf v; };

__device__ __forceinline__ unsigned short f2bf(float f) {
  unsigned u = __float_as_uint(f);
  u += 0x7FFFu + ((u >> 16) & 1u);      // round-to-nearest-even
  return (unsigned short)(u >> 16);
}

// async copy 16B global -> LDS (CDNA5, ASYNCcnt-tracked)
__device__ __forceinline__ void async_copy_b128(unsigned lds_addr,
                                                const unsigned short* gptr) {
  asm volatile("global_load_async_to_lds_b128 %0, %1, off"
               :: "v"(lds_addr), "v"(gptr) : "memory");
}
__device__ __forceinline__ void wait_asynccnt0() {
  asm volatile("s_wait_asynccnt 0x0" ::: "memory");
}

// ---------------------------------------------------------------------------
// fp32 -> bf16 elementwise convert
// ---------------------------------------------------------------------------
__global__ void cvt_f32_bf16(const float* __restrict__ in,
                             unsigned short* __restrict__ out, int n) {
  int i = blockIdx.x * blockDim.x + threadIdx.x;
  if (i < n) out[i] = f2bf(in[i]);
}

// fp32 [K][N] -> bf16 [N][K]  (transpose so B-fragment loads are contiguous)
__global__ void transpose_cvt(const float* __restrict__ in,
                              unsigned short* __restrict__ out, int K, int Nn) {
  int k = blockIdx.x * 16 + threadIdx.x;
  int n = blockIdx.y * 16 + threadIdx.y;
  if (k < K && n < Nn) out[(size_t)n * K + k] = f2bf(in[(size_t)k * Nn + n]);
}

// ---------------------------------------------------------------------------
// bf16 GEMM, LDS double-buffered with async global->LDS staging.
//   C[M][Nn] = A[M][K] * Bt[Nn][K]^T + bias     (f32 accum via WMMA)
// Block = 256 threads (8 waves). Block tile 64(M) x 256(N), K-step 64.
// Wave (wm in 0..1, wn in 0..3) owns a 32(M) x 64(N) set of 16x16 C tiles
// (2x4), so A frags are reused 4x and B frags 2x; 16 WMMA issued per K-step.
// LDS rows padded to 72 halfs (144B) -> conflict-free ds_load_b128.
// ---------------------------------------------------------------------------
template <bool RELU, bool STORE_F32, bool STORE_BF16>
__global__ __launch_bounds__(256) void gemm_bf16_wmma(
    const unsigned short* __restrict__ A,    // [M][K]  bf16 raw
    const unsigned short* __restrict__ Bt,   // [Nn][K] bf16 raw (transposed)
    const float* __restrict__ bias,          // [Nn]
    float* __restrict__ Cf,                  // [M][Nn] (optional)
    unsigned short* __restrict__ Cb,         // [M][Nn] bf16 (optional)
    int M, int Nn, int K) {
  __shared__ __align__(16) unsigned short sA[2][64][APAD];
  __shared__ __align__(16) unsigned short sB[2][256][APAD];

  const int lane  = threadIdx.x & 31;
  const int wave  = threadIdx.x >> 5;
  const int wm    = wave >> 2;               // 0..1
  const int wn    = wave & 3;                // 0..3
  const int m0g   = blockIdx.y * 64;
  const int n0g   = blockIdx.x * 256;
  const int lhalf = lane & 15;
  const int hi    = lane >> 4;               // 0 = lanes 0-15, 1 = lanes 16-31

  // issue async staging of one K-step tile (16B units; A: 512, B: 2048)
  auto issue_tile = [&](int buf, int kb) {
#pragma unroll
    for (int j = 0; j < 2; ++j) {            // A tile: 64 rows x 64 halfs
      const int u   = threadIdx.x + j * 256;
      const int row = u >> 3;
      const int col = (u & 7) * 8;
      async_copy_b128((unsigned)(uintptr_t)&sA[buf][row][col],
                      A + (size_t)(m0g + row) * K + kb + col);
    }
#pragma unroll
    for (int j = 0; j < 8; ++j) {            // B tile: 256 rows x 64 halfs
      const int u   = threadIdx.x + j * 256;
      const int row = u >> 3;
      const int col = (u & 7) * 8;
      async_copy_b128((unsigned)(uintptr_t)&sB[buf][row][col],
                      Bt + (size_t)(n0g + row) * K + kb + col);
    }
  };

  v8f acc[2][4] = {};

  issue_tile(0, 0);

  for (int kb = 0; kb < K; kb += 64) {
    const int cur = (kb >> 6) & 1;
    wait_asynccnt0();                        // our staging for buf `cur` landed
    __syncthreads();                         // everyone's staging landed
    if (kb + 64 < K) issue_tile(cur ^ 1, kb + 64);  // overlap next tile

#pragma unroll
    for (int kc = 0; kc < 64; kc += 32) {
      FragBF fa[2], fb[4];
      // A frag (ISA 16-bit A layout): lo half-wave K {0..7,16..23},
      // hi half-wave K {8..15,24..31} within the 32-chunk
#pragma unroll
      for (int mt = 0; mt < 2; ++mt) {
        const unsigned short* p = &sA[cur][wm * 32 + mt * 16 + lhalf][kc + hi * 8];
        fa[mt].q[0] = *(const uint4*)(p);
        fa[mt].q[1] = *(const uint4*)(p + 16);
      }
      // B frag: lane holds column n, K {0..7,16..23} (hi half-wave {8..15,24..31})
#pragma unroll
      for (int nt = 0; nt < 4; ++nt) {
        const unsigned short* p = &sB[cur][wn * 64 + nt * 16 + lhalf][kc + hi * 8];
        fb[nt].q[0] = *(const uint4*)(p);
        fb[nt].q[1] = *(const uint4*)(p + 16);
      }
#pragma unroll
      for (int mt = 0; mt < 2; ++mt)
#pragma unroll
        for (int nt = 0; nt < 4; ++nt)
          acc[mt][nt] = __builtin_amdgcn_wmma_f32_16x16x32_bf16(
              false, fa[mt].v, false, fb[nt].v, (short)0, acc[mt][nt],
              false, false);
    }
    __syncthreads();                         // done reading buf `cur`
  }

  // Epilogue: C vgpr r -> row base + r + 8*hi, col = base + lhalf
#pragma unroll
  for (int mt = 0; mt < 2; ++mt) {
#pragma unroll
    for (int nt = 0; nt < 4; ++nt) {
      const int col = n0g + wn * 64 + nt * 16 + lhalf;
      const float bv = bias[col];
#pragma unroll
      for (int r = 0; r < 8; ++r) {
        const int row = m0g + wm * 32 + mt * 16 + r + hi * 8;
        float val = acc[mt][nt][r] + bv;
        if (RELU) val = val > 0.f ? val : 0.f;
        const size_t o = (size_t)row * Nn + col;
        if (STORE_F32)  Cf[o] = val;
        if (STORE_BF16) Cb[o] = f2bf(val);
      }
    }
  }
}

// ---------------------------------------------------------------------------
// Per-token attention over the HEAD axis (16x16x64): one wave per token.
// scores = q k^T / 8 via V_WMMA_F32_16X16X4_F32, softmax in C-layout with
// 16-lane shfl reductions, attn*V via LDS (negligible FLOPs).
// qkv layout per token: [head][192] with q=0..63, k=64..127, v=128..191.
// ---------------------------------------------------------------------------
__global__ __launch_bounds__(128) void attention_kernel(
    const float* __restrict__ qkv,           // [NTOK][16][192]
    unsigned short* __restrict__ ctx_b) {    // [NTOK][1024] bf16
  __shared__ float s_attn[4][16][16];
  __shared__ float s_v[4][16][64];
  const int lane  = threadIdx.x & 31;
  const int w     = threadIdx.x >> 5;
  const int token = blockIdx.x * 4 + w;
  const float* base = qkv + (size_t)token * (NHEADS * 3 * DK);

  // stage V into LDS
  for (int idx = lane; idx < NHEADS * DK; idx += 32) {
    int g = idx >> 6, d = idx & 63;
    s_v[w][g][d] = base[g * 192 + 128 + d];
  }

  const int lhalf = lane & 15;
  const int hi    = lane >> 4;

  // scores: A = q (16x64), B = k^T (64x16). f32 A/B frags are v2f per lane:
  // lo half-wave K {k0,k0+1}, hi half-wave K {k0+2,k0+3}.
  v8f c = {};
#pragma unroll
  for (int k0 = 0; k0 < DK; k0 += 4) {
    const float* pq = base + lhalf * 192 + k0 + hi * 2;        // q[m][..]
    const float* pk = base + lhalf * 192 + 64 + k0 + hi * 2;   // k[n][..]
    v2f a; a.x = pq[0]; a.y = pq[1];
    v2f b; b.x = pk[0]; b.y = pk[1];
    c = __builtin_amdgcn_wmma_f32_16x16x4_f32(
        false, a, false, b, (short)0, c, false, false);
  }

  // scale + row softmax (row = r + 8*hi lives across the 16 lanes of a half-wave)
#pragma unroll
  for (int r = 0; r < 8; ++r) {
    float v0 = c[r] * 0.125f;                // 1/sqrt(64)
    float mx = v0;
#pragma unroll
    for (int m = 1; m < 16; m <<= 1) mx = fmaxf(mx, __shfl_xor(mx, m, 16));
    float e = __expf(v0 - mx);
    float sm = e;
#pragma unroll
    for (int m = 1; m < 16; m <<= 1) sm += __shfl_xor(sm, m, 16);
    c[r] = e / sm;
  }
#pragma unroll
  for (int r = 0; r < 8; ++r) s_attn[w][r + 8 * hi][lhalf] = c[r];
  __syncthreads();

  // context[h][d] = sum_g attn[h][g] * v[g][d]
#pragma unroll
  for (int h = 0; h < NHEADS; ++h) {
#pragma unroll
    for (int j = 0; j < 2; ++j) {
      const int d = lane + 32 * j;
      float s = 0.f;
#pragma unroll
      for (int g = 0; g < 16; ++g) s += s_attn[w][h][g] * s_v[w][g][d];
      ctx_b[(size_t)token * DMODEL + h * DK + d] = f2bf(s);
    }
  }
}

// ---------------------------------------------------------------------------
// y = LayerNorm(a + b) * gamma + beta ; optional bf16 copy for next GEMM
// ---------------------------------------------------------------------------
template <bool WRITE_BF16>
__global__ __launch_bounds__(256) void ln_residual_kernel(
    const float* __restrict__ a, const float* __restrict__ b,
    const float* __restrict__ gamma, const float* __restrict__ beta,
    float* __restrict__ outf, unsigned short* __restrict__ outb) {
  __shared__ float red[256];
  __shared__ float red2[256];
  const int t = threadIdx.x;
  const size_t base = (size_t)blockIdx.x * DMODEL;
  float v[4];
  float s = 0.f, s2 = 0.f;
#pragma unroll
  for (int i = 0; i < 4; ++i) {
    const int idx = t + i * 256;
    const float x = a[base + idx] + b[base + idx];
    v[i] = x; s += x; s2 += x * x;
  }
  red[t] = s; red2[t] = s2;
  __syncthreads();
  for (int off = 128; off > 0; off >>= 1) {
    if (t < off) { red[t] += red[t + off]; red2[t] += red2[t + off]; }
    __syncthreads();
  }
  const float mu   = red[0] * (1.f / DMODEL);
  const float var  = red2[0] * (1.f / DMODEL) - mu * mu;
  const float rstd = rsqrtf(var + LN_EPS);
#pragma unroll
  for (int i = 0; i < 4; ++i) {
    const int idx = t + i * 256;
    const float y = (v[i] - mu) * rstd * gamma[idx] + beta[idx];
    outf[base + idx] = y;
    if (WRITE_BF16) outb[base + idx] = f2bf(y);
  }
}

// ---------------------------------------------------------------------------
extern "C" void kernel_launch(void* const* d_in, const int* in_sizes, int n_in,
                              void* d_out, int out_size, void* d_ws, size_t ws_size,
                              hipStream_t stream) {
  (void)in_sizes; (void)n_in; (void)out_size; (void)ws_size;
  const float* x    = (const float*)d_in[0];
  const float* Wqkv = (const float*)d_in[1];
  const float* bqkv = (const float*)d_in[2];
  const float* Wo   = (const float*)d_in[3];
  const float* bo   = (const float*)d_in[4];
  const float* W1   = (const float*)d_in[5];
  const float* b1   = (const float*)d_in[6];
  const float* W2   = (const float*)d_in[7];
  const float* b2   = (const float*)d_in[8];
  const float* g1   = (const float*)d_in[9];
  const float* be1  = (const float*)d_in[10];
  const float* g2   = (const float*)d_in[11];
  const float* be2  = (const float*)d_in[12];

  char* ws = (char*)d_ws;
  size_t off = 0;
  auto alloc = [&](size_t bytes) -> char* {
    char* p = ws + off;
    off += (bytes + 255) & ~(size_t)255;
    return p;
  };

  unsigned short* xb     = (unsigned short*)alloc((size_t)NTOK * DMODEL * 2);
  unsigned short* wqkvT  = (unsigned short*)alloc((size_t)3 * DMODEL * DMODEL * 2);
  unsigned short* woT    = (unsigned short*)alloc((size_t)DMODEL * DMODEL * 2);
  unsigned short* w1T    = (unsigned short*)alloc((size_t)DFF * DMODEL * 2);
  unsigned short* w2T    = (unsigned short*)alloc((size_t)DMODEL * DFF * 2);
  float*          qkvf   = (float*)alloc((size_t)NTOK * 3 * DMODEL * 4);
  unsigned short* ctxb   = (unsigned short*)alloc((size_t)NTOK * DMODEL * 2);
  float*          attnf  = (float*)alloc((size_t)NTOK * DMODEL * 4);
  float*          x1f    = (float*)alloc((size_t)NTOK * DMODEL * 4);
  unsigned short* x1b    = (unsigned short*)alloc((size_t)NTOK * DMODEL * 2);
  unsigned short* hb     = (unsigned short*)alloc((size_t)NTOK * DFF * 2);
  float*          fff    = (float*)alloc((size_t)NTOK * DMODEL * 4);

  // 1) convert activations / weights to bf16 (weights transposed to [N][K])
  {
    const int n = NTOK * DMODEL;
    cvt_f32_bf16<<<n / 256, 256, 0, stream>>>(x, xb, n);
  }
  transpose_cvt<<<dim3(DMODEL / 16, 3 * DMODEL / 16), dim3(16, 16), 0, stream>>>(
      Wqkv, wqkvT, DMODEL, 3 * DMODEL);
  transpose_cvt<<<dim3(DMODEL / 16, DMODEL / 16), dim3(16, 16), 0, stream>>>(
      Wo, woT, DMODEL, DMODEL);
  transpose_cvt<<<dim3(DMODEL / 16, DFF / 16), dim3(16, 16), 0, stream>>>(
      W1, w1T, DMODEL, DFF);
  transpose_cvt<<<dim3(DFF / 16, DMODEL / 16), dim3(16, 16), 0, stream>>>(
      W2, w2T, DFF, DMODEL);

  // 2) QKV projection: [NTOK,1024] x [1024,3072]
  gemm_bf16_wmma<false, true, false>
      <<<dim3(3 * DMODEL / 256, NTOK / 64), 256, 0, stream>>>(
          xb, wqkvT, bqkv, qkvf, nullptr, NTOK, 3 * DMODEL, DMODEL);

  // 3) per-token attention over heads
  attention_kernel<<<NTOK / 4, 128, 0, stream>>>(qkvf, ctxb);

  // 4) output projection
  gemm_bf16_wmma<false, true, false>
      <<<dim3(DMODEL / 256, NTOK / 64), 256, 0, stream>>>(
          ctxb, woT, bo, attnf, nullptr, NTOK, DMODEL, DMODEL);

  // 5) x1 = LN(x + attn_out)
  ln_residual_kernel<true><<<NTOK, 256, 0, stream>>>(x, attnf, g1, be1, x1f, x1b);

  // 6) h = relu(x1 @ W1 + b1) -> bf16
  gemm_bf16_wmma<true, false, true>
      <<<dim3(DFF / 256, NTOK / 64), 256, 0, stream>>>(
          x1b, w1T, b1, nullptr, hb, NTOK, DFF, DMODEL);

  // 7) ff = h @ W2 + b2
  gemm_bf16_wmma<false, true, false>
      <<<dim3(DMODEL / 256, NTOK / 64), 256, 0, stream>>>(
          hb, w2T, b2, fff, nullptr, NTOK, DMODEL, DFF);

  // 8) out = LN(x1 + ff)
  ln_residual_kernel<false><<<NTOK, 256, 0, stream>>>(
      x1f, fff, g2, be2, (float*)d_out, nullptr);
}
